// MixtralMoE_47949014893023
// MI455X (gfx1250) — matrix operand order repaired and checked by reading, not compiled, not run on hardware
//
#include <hip/hip_runtime.h>

// ---------------------------------------------------------------------------
// Mixtral MoE layer for gfx1250 (MI455X): token-gathered expert GEMMs using
// v_wmma_f32_16x16x32_bf16 (2-way M register tiling, v_cvt_pk_bf16_f32
// conversions), Tensor Data Mover staging for the bf16 act panel,
// L2-resident weight panels, deterministic top-2 scatter-add epilogue.
// ---------------------------------------------------------------------------

static constexpr int Td = 1024;   // tokens
static constexpr int Hd = 1024;   // hidden
static constexpr int Id = 2048;   // intermediate
static constexpr int Ed = 8;      // experts

typedef __attribute__((ext_vector_type(16))) __bf16 v16bf;
typedef __attribute__((ext_vector_type(8)))  __bf16 v8bf;
typedef __attribute__((ext_vector_type(4)))  __bf16 v4bf;
typedef __attribute__((ext_vector_type(8)))  float  v8f;
typedef __attribute__((ext_vector_type(4)))  float  v4f;
typedef __attribute__((ext_vector_type(4)))  unsigned int v4u;
typedef __attribute__((ext_vector_type(8)))  int    v8i;
typedef __attribute__((ext_vector_type(4)))  int    v4i;

__device__ __forceinline__ __bf16 f2bf(float f) { return (__bf16)f; }

// load 16 contiguous fp32, convert to bf16 fragment (B operand: lane holds
// col = lane&15, k = j + (lane>=16 ? 16 : 0), contiguous run of 16)
__device__ __forceinline__ v16bf cvt16(const float* __restrict__ p) {
  v16bf r;
#pragma unroll
  for (int j = 0; j < 16; ++j) r[j] = (__bf16)p[j];
  return r;
}

// A-fragment from an LDS bf16 row: elements j at [kk+akoff+j],
// contiguous run of 16 (matches B fragment k order)
__device__ __forceinline__ v16bf lds_a_frag(const __bf16* __restrict__ row,
                                            int kk, int akoff) {
  v8bf lo = *(const v8bf*)(row + kk + akoff);
  v8bf hi = *(const v8bf*)(row + kk + akoff + 8);
  return __builtin_shufflevector(lo, hi, 0, 1, 2, 3, 4, 5, 6, 7,
                                 8, 9, 10, 11, 12, 13, 14, 15);
}

__device__ __forceinline__ v8f wmma_bf16(v16bf a, v16bf b, v8f c) {
  return __builtin_amdgcn_wmma_f32_16x16x32_bf16(
      /*neg_a=*/false, a, /*neg_b=*/false, b,
      /*c_mod=*/(short)0, c, /*reuse_a=*/false, /*reuse_b=*/false);
}

// silu(h1)*h3 with fast reciprocal (v_rcp_f32, ~1ulp; output is bf16)
__device__ __forceinline__ float silu_mul(float h1, float h3) {
  float s = __builtin_amdgcn_rcpf(1.0f + __expf(-h1));
  return h1 * s * h3;
}

// ---------------------------------------------------------------------------
// Kernel 0: zero output + expert counters (d_out / d_ws arrive poisoned)
// ---------------------------------------------------------------------------
__global__ __launch_bounds__(256) void moe_init(float* __restrict__ out,
                                                int* __restrict__ cnt) {
  int i = blockIdx.x * blockDim.x + threadIdx.x;
  if (i < Td * Hd) out[i] = 0.0f;
  if (i < Ed) cnt[i] = 0;
}

// ---------------------------------------------------------------------------
// Kernel 1: router. One wave32 per token: logits, softmax, top-2, renorm,
// compacted per-expert token lists via atomic counters.
// ---------------------------------------------------------------------------
__global__ __launch_bounds__(256) void moe_router(
    const int* __restrict__ d_index, const float* __restrict__ x,
    const float* __restrict__ gate_w, int* __restrict__ cnt,
    int* __restrict__ tok_list, float* __restrict__ tok_w) {
  const int lane = threadIdx.x & 31;
  const int wave = threadIdx.x >> 5;
  const int t = blockIdx.x * 8 + wave;
  const float* gate = gate_w + (size_t)(*d_index) * Ed * Hd;

  float lg[Ed];
#pragma unroll
  for (int e = 0; e < Ed; ++e) lg[e] = 0.0f;

  for (int h = lane; h < Hd; h += 32) {
    float xv = x[(size_t)t * Hd + h];
#pragma unroll
    for (int e = 0; e < Ed; ++e) lg[e] += xv * gate[e * Hd + h];
  }
#pragma unroll
  for (int off = 16; off > 0; off >>= 1) {
#pragma unroll
    for (int e = 0; e < Ed; ++e) lg[e] += __shfl_xor(lg[e], off, 32);
  }

  if (lane == 0) {
    float mx = lg[0];
#pragma unroll
    for (int e = 1; e < Ed; ++e) mx = fmaxf(mx, lg[e]);
    float pe[Ed];
#pragma unroll
    for (int e = 0; e < Ed; ++e) pe[e] = __expf(lg[e] - mx);
    // softmax denominator cancels in the top-2 renormalization
    int i1 = 0;
#pragma unroll
    for (int e = 1; e < Ed; ++e)
      if (pe[e] > pe[i1]) i1 = e;          // strict > keeps lowest index
    int i2 = (i1 == 0) ? 1 : 0;
#pragma unroll
    for (int e = 0; e < Ed; ++e)
      if (e != i2 && e != i1 && pe[e] > pe[i2]) i2 = e;
    float inv = 1.0f / (pe[i1] + pe[i2]);

    int p1 = atomicAdd(&cnt[i1], 1);
    tok_list[i1 * Td + p1] = t;
    tok_w[i1 * Td + p1] = pe[i1] * inv;
    int p2 = atomicAdd(&cnt[i2], 1);
    tok_list[i2 * Td + p2] = t;
    tok_w[i2 * Td + p2] = pe[i2] * inv;
  }
}

// ---------------------------------------------------------------------------
// Kernel 2: gate/up projections + SiLU.  grid = E * 32 * 16 (expert-major),
// block = 256 (8 waves). Each block: 32 gathered tokens x 128 cols of I.
// A panel (32 x 1024 bf16, 64KB) staged once in LDS (manual: fuses fp32->bf16
// conversion). Each wave: two 16x16 M-tiles sharing each converted B fragment.
// ---------------------------------------------------------------------------
__global__ __launch_bounds__(256) void moe_ffn1(
    const int* __restrict__ d_index, const float* __restrict__ x,
    const float* __restrict__ ws, const int* __restrict__ cnt,
    const int* __restrict__ tok_list, __bf16* __restrict__ act) {
  const int e = blockIdx.x >> 9;           // 32*16 blocks per expert
  const int rem = blockIdx.x & 511;
  const int mg = rem >> 4;                 // group of two 16-token tiles
  const int nblk = rem & 15;               // 128-col chunk of I
  const int n_e = cnt[e];
  if (mg * 32 >= n_e) return;

  __shared__ __bf16 lsA[32][Hd];           // 64 KB

  // stage gathered token panel (rows past n_e duplicate the first token; they
  // produce finite garbage that kernel 3 never emits). Packed 8B DS stores.
  {
    const int r = threadIdx.x >> 3;        // 0..31
    const int cb = (threadIdx.x & 7) * 128;
    const int slot = mg * 32 + r;
    const int tok = tok_list[e * Td + ((slot < n_e) ? slot : 0)];
    const float* src = x + (size_t)tok * Hd + cb;
#pragma unroll
    for (int j = 0; j < 32; ++j) {
      v4f v = ((const v4f*)src)[j];
      v4bf o;
#pragma unroll
      for (int q = 0; q < 4; ++q) o[q] = (__bf16)v[q];
      ((v4bf*)&lsA[r][cb])[j] = o;
    }
  }
  __syncthreads();

  const int lane = threadIdx.x & 31;
  const int wave = threadIdx.x >> 5;
  const int c = nblk * 128 + wave * 16 + (lane & 15);  // column in I
  const int akoff = (lane >= 16) ? 16 : 0;
  const int bkoff = (lane >= 16) ? 16 : 0;
  const int arow = lane & 15;

  const float* wl = ws + (size_t)(*d_index) * Ed * 3 * Id * Hd;
  const float* b1p = wl + (size_t)e * 3 * Id * Hd + (size_t)c * Hd;  // w1 row c
  const float* b3p = b1p + (size_t)Id * Hd;                          // w3 row c

  v8f acc1[2] = {v8f{}, v8f{}};
  v8f acc3[2] = {v8f{}, v8f{}};
  for (int kk = 0; kk < Hd; kk += 32) {
    v16bf b1 = cvt16(b1p + kk + bkoff);
    v16bf b3 = cvt16(b3p + kk + bkoff);
#pragma unroll
    for (int mt = 0; mt < 2; ++mt) {
      v16bf a = lds_a_frag(&lsA[mt * 16 + arow][0], kk, akoff);
      acc1[mt] = wmma_bf16(a, b1, acc1[mt]);
      acc3[mt] = wmma_bf16(a, b3, acc3[mt]);
    }
  }

  // SiLU(h1) * h3 -> act[(e*T + slot) * I + c], bf16
  const int mrow = (lane >= 16) ? 8 : 0;
#pragma unroll
  for (int mt = 0; mt < 2; ++mt) {
#pragma unroll
    for (int r = 0; r < 8; ++r) {
      const int slot = mg * 32 + mt * 16 + r + mrow;
      act[((size_t)e * Td + slot) * Id + c] =
          f2bf(silu_mul(acc1[mt][r], acc3[mt][r]));
    }
  }
}

// ---------------------------------------------------------------------------
// Kernel 3: down projection + weighted scatter-add.  grid = E * 32 * 8,
// block = 256. 32-row act panel staged per 64KB half via the Tensor Data
// Mover (2D D#: 1024x32 bf16 tile, row stride I). Wave 0 issues the DMA,
// waits s_wait_tensorcnt 0, then a workgroup barrier publishes the panel.
// Epilogue: out[token, h] += route_w * acc via fp32 atomics (K=2 -> exact
// commutativity -> deterministic).
// ---------------------------------------------------------------------------
__global__ __launch_bounds__(256) void moe_ffn2(
    const int* __restrict__ d_index, const float* __restrict__ ws,
    const int* __restrict__ cnt, const int* __restrict__ tok_list,
    const float* __restrict__ tok_w, const __bf16* __restrict__ act,
    float* __restrict__ out) {
  const int e = blockIdx.x >> 8;           // 32*8 blocks per expert
  const int rem = blockIdx.x & 255;
  const int mg = rem >> 3;                 // group of two 16-row tiles
  const int nblk = rem & 7;                // 128-col chunk of H
  const int n_e = cnt[e];
  if (mg * 32 >= n_e) return;

  __shared__ __bf16 lsA[32][1024];         // 64 KB, half of the I panel

  const int lane = threadIdx.x & 31;
  const int wave = threadIdx.x >> 5;
  const int h = nblk * 128 + wave * 16 + (lane & 15);  // output column in H
  const int akoff = (lane >= 16) ? 16 : 0;
  const int bkoff = (lane >= 16) ? 16 : 0;
  const int arow = lane & 15;

  const float* wl = ws + (size_t)(*d_index) * Ed * 3 * Id * Hd;
  const float* b2p =
      wl + (size_t)e * 3 * Id * Hd + 2 * (size_t)Id * Hd + (size_t)h * Id;

  v8f accd[2] = {v8f{}, v8f{}};
  for (int half = 0; half < 2; ++half) {
    const __bf16* tile_src =
        act + ((size_t)e * Td + mg * 32) * Id + half * 1024;
    if (wave == 0) {
      // Tensor DMA descriptor (D#), normal (non-gather) mode, 2D tile:
      //   data_size = 2B, tile_dim0 = 1024 cols, tile_dim1 = 32 rows,
      //   tensor_dim0 = I (row length), tensor_dim0_stride = I.
      unsigned long long ga = (unsigned long long)(uintptr_t)tile_src;
      unsigned int ldsoff = (unsigned int)(uintptr_t)(&lsA[0][0]);
      v4u g0;
      g0[0] = 1u;                                        // count=1, user mode
      g0[1] = ldsoff;                                    // lds_addr
      g0[2] = (unsigned int)(ga & 0xFFFFFFFFu);          // global_addr[31:0]
      g0[3] = (unsigned int)((ga >> 32) & 0x01FFFFFFu)   // global_addr[56:32]
              | (2u << 30);                              // type=2 ("image")
      v8i g1;
      g1[0] = 0x00010000;                 // wg_mask=0, data_size=1 (2 bytes)
      g1[1] = (int)((unsigned)Id << 16);  // tensor_dim0[15:0]=2048 @ [63:48]
      g1[2] = (int)(32u << 16);           // tensor_dim0 hi=0; tensor_dim1=32
      g1[3] = (int)(1024u << 16);         // tensor_dim1 hi=0; tile_dim0=1024
      g1[4] = 32;                         // tile_dim1=32, tile_dim2=0
      g1[5] = Id;                         // tensor_dim0_stride[31:0] = 2048
      g1[6] = 0;                          // stride hi, dim1_stride lo
      g1[7] = 0;
      v4i gz4 = {0, 0, 0, 0};             // groups 2/3 unused (2D tensor)
      v8i gz8 = {0, 0, 0, 0, 0, 0, 0, 0}; // trailing group (clang-23 form)
      __builtin_amdgcn_tensor_load_to_lds(g0, g1, gz4, gz4, gz8, 0);
      __builtin_amdgcn_s_wait_tensorcnt(0);
    }
    __syncthreads();

    const float* bp = b2p + half * 1024;
    for (int kk = 0; kk < 1024; kk += 32) {
      v16bf b = cvt16(bp + kk + bkoff);
#pragma unroll
      for (int mt = 0; mt < 2; ++mt) {
        v16bf a = lds_a_frag(&lsA[mt * 16 + arow][0], kk, akoff);
        accd[mt] = wmma_bf16(a, b, accd[mt]);
      }
    }
    __syncthreads();
  }

  const int mrow = (lane >= 16) ? 8 : 0;
#pragma unroll
  for (int mt = 0; mt < 2; ++mt) {
#pragma unroll
    for (int r = 0; r < 8; ++r) {
      const int slot = mg * 32 + mt * 16 + r + mrow;
      if (slot < n_e) {
        const int tok = tok_list[e * Td + slot];
        const float w = tok_w[e * Td + slot];
        atomicAdd(&out[(size_t)tok * Hd + h], w * accd[mt][r]);
      }
    }
  }
}

// ---------------------------------------------------------------------------
extern "C" void kernel_launch(void* const* d_in, const int* in_sizes, int n_in,
                              void* d_out, int out_size, void* d_ws,
                              size_t ws_size, hipStream_t stream) {
  const int* d_index = (const int*)d_in[0];     // scalar layer index
  const float* x = (const float*)d_in[1];       // [T, H]
  // d_in[2]: experts_cache (unused)
  const float* gate_w = (const float*)d_in[3];  // [L, E, H]
  const float* ws = (const float*)d_in[4];      // [L, E, 3*I*H]
  float* out = (float*)d_out;                   // [T, H]

  char* wsb = (char*)d_ws;
  int* cnt = (int*)wsb;                               // E ints
  int* tok_list = (int*)(wsb + 1024);                 // E*T ints
  float* tok_w = (float*)(wsb + 1024 + Ed * Td * 4);  // E*T floats
  __bf16* act = (__bf16*)(wsb + 1024 + 2 * Ed * Td * 4);  // E*T*I bf16 (~32MB)

  moe_init<<<(Td * Hd + 255) / 256, 256, 0, stream>>>(out, cnt);
  moe_router<<<Td / 8, 256, 0, stream>>>(d_index, x, gate_w, cnt, tok_list,
                                         tok_w);
  moe_ffn1<<<Ed * 32 * 16, 256, 0, stream>>>(d_index, x, ws, cnt, tok_list,
                                             act);
  moe_ffn2<<<Ed * 32 * 8, 256, 0, stream>>>(d_index, ws, cnt, tok_list, tok_w,
                                            act, out);
}
